// SpatialSelfAttention_35175782154351
// MI455X (gfx1250) — matrix operand order repaired and checked by reading, not compiled, not run on hardware
//
#include <hip/hip_runtime.h>

// ---------------------------------------------------------------------------
// Types for CDNA5 WMMA (wave32): v_wmma_f32_16x16x32_bf16
// ---------------------------------------------------------------------------
typedef __bf16 bf16_t;
typedef bf16_t v16bf __attribute__((ext_vector_type(16)));
typedef float v8f __attribute__((ext_vector_type(8)));
typedef unsigned short u16;
typedef u16 v8u __attribute__((ext_vector_type(8)));
typedef u16 v16u __attribute__((ext_vector_type(16)));

#define DEV __device__ __forceinline__

DEV u16 f32_to_bf16(float f) {
    unsigned int u = __builtin_bit_cast(unsigned int, f);
    u += 0x7FFFu + ((u >> 16) & 1u);
    return (u16)(u >> 16);
}

DEV v16bf combine16(v8u a, v8u b) {
    v16u t = __builtin_shufflevector(a, b, 0, 1, 2, 3, 4, 5, 6, 7,
                                     8, 9, 10, 11, 12, 13, 14, 15);
    return __builtin_bit_cast(v16bf, t);
}

// A fragment (16x32 bf16, M x K). Lane L: row = L%16, K pairs per ISA layout:
// elements 0..7 = K[koff..koff+7], 8..15 = K[koff+16..koff+23], koff=(L/16)*8.
DEV v16bf load_a_frag(const u16* p, int pitch) {
    int lane = threadIdx.x & 31;
    const u16* row = p + (size_t)(lane & 15) * pitch + ((lane >> 4) << 3);
    v8u lo = *(const v8u*)(row);
    v8u hi = *(const v8u*)(row + 16);
    return combine16(lo, hi);
}

// B fragment (32x16 bf16, K x N), operand given as Bt[n][k] (K contiguous).
// Lane L: col = L%16, elements 0..7 = K[koff..koff+7], 8..15 = K[koff+16..koff+23], koff=(L/16)*8.
DEV v16bf load_b_frag(const u16* p, int pitch) {
    int lane = threadIdx.x & 31;
    const u16* row = p + (size_t)(lane & 15) * pitch + ((lane >> 4) << 3);
    v8u lo = *(const v8u*)(row);
    v8u hi = *(const v8u*)(row + 16);
    return combine16(lo, hi);
}

DEV v8f wmma_bf16(v16bf a, v16bf b, v8f c) {
    return __builtin_amdgcn_wmma_f32_16x16x32_bf16(
        false, a, false, b, (short)0, c, false, false);
}

// Problem constants
#define CC 512
#define NTOK 4096
#define BATCH 4
#define MTOT (BATCH * NTOK)   // 16384

// ---------------------------------------------------------------------------
// Kernel 1: f32 -> bf16 weight conversion (weights are already "Bt" layout)
// ---------------------------------------------------------------------------
__global__ __launch_bounds__(256) void cvt_kernel(const float* __restrict__ src,
                                                  u16* __restrict__ dst, int n) {
    int i = blockIdx.x * 256 + threadIdx.x;
    if (i < n) dst[i] = f32_to_bf16(src[i]);
}

// ---------------------------------------------------------------------------
// Kernel 2: GroupNorm (32 groups of 16 ch) -> h_t bf16 token-major [B*N, C]
// ---------------------------------------------------------------------------
__global__ __launch_bounds__(256) void gn_kernel(const float* __restrict__ x,
                                                 const float* __restrict__ gs,
                                                 const float* __restrict__ gb,
                                                 u16* __restrict__ h_t) {
    __shared__ float red[256];
    __shared__ float red2[256];
    __shared__ float s_mean, s_rstd;
    __shared__ float sc[16], sb[16];

    int bg = blockIdx.x;
    int b = bg >> 5, g = bg & 31;
    int c0 = g * 16;
    const float* xp = x + ((size_t)b * CC + c0) * NTOK;
    int tid = threadIdx.x;

    float s = 0.f, s2 = 0.f;
    for (int i = tid; i < 16 * NTOK; i += 256) {
        float v = xp[i];
        s += v;
        s2 += v * v;
    }
    red[tid] = s;
    red2[tid] = s2;
    __syncthreads();
    for (int off = 128; off > 0; off >>= 1) {
        if (tid < off) {
            red[tid] += red[tid + off];
            red2[tid] += red2[tid + off];
        }
        __syncthreads();
    }
    if (tid == 0) {
        float mean = red[0] * (1.f / 65536.f);
        float var = red2[0] * (1.f / 65536.f) - mean * mean;
        s_mean = mean;
        s_rstd = rsqrtf(var + 1e-6f);
    }
    __syncthreads();
    if (tid < 16) {
        float a = gs[c0 + tid] * s_rstd;
        sc[tid] = a;
        sb[tid] = gb[c0 + tid] - a * s_mean;
    }
    __syncthreads();

    for (int nb = 0; nb < NTOK; nb += 256) {
        int n = nb + tid;
        v8u p0, p1;
#pragma unroll
        for (int c = 0; c < 8; ++c)
            p0[c] = f32_to_bf16(xp[(size_t)c * NTOK + n] * sc[c] + sb[c]);
#pragma unroll
        for (int c = 0; c < 8; ++c)
            p1[c] = f32_to_bf16(xp[(size_t)(c + 8) * NTOK + n] * sc[c + 8] + sb[c + 8]);
        u16* dst = h_t + (size_t)(b * NTOK + n) * CC + c0;
        *(v8u*)(dst) = p0;
        *(v8u*)(dst + 8) = p1;
    }
}

// ---------------------------------------------------------------------------
// Kernel 3: TN GEMM  D[m][n] = sum_k A[m][k] * Bt[n][k], K = 512.
// Block tile 128x64, 8 waves, wave tile 32x32 (4 WMMAs per K-step).
// Double-buffered LDS staging: stage chunk k+1 while computing chunk k,
// one barrier per K-step.
// MODE 0: out bf16 token-major [MTOT, 512]:  out = (acc + bias) * alpha (q/k)
// MODE 1: out bf16 transposed  [B][C][N]:    v_ct (coalesced 16B stores)
// MODE 2: out f32 channel-major [B][C][N]:   residual + bias (final proj)
// ---------------------------------------------------------------------------
template <int MODE>
__global__ __launch_bounds__(256) void gemm_tn_kernel(
    const u16* __restrict__ A, const u16* __restrict__ Bt,
    const float* __restrict__ bias, const float* __restrict__ resid,
    float alpha, u16* __restrict__ out_b, float* __restrict__ out_f) {
    __shared__ u16 a_lds[2][128][40];
    __shared__ u16 b_lds[2][64][40];

    int m0 = blockIdx.x * 128;
    int n0 = blockIdx.y * 64;
    int tid = threadIdx.x;
    int lane = tid & 31;
    int w = tid >> 5;
    int mt2 = w >> 1, nb = w & 1;  // wave tile: rows mt2*32, cols nb*32

    // staging map: A 128 rows x 32 (2x v8u per thread), B 64 rows x 32 (1x v8u)
    int ar = tid >> 1, ac = (tid & 1) * 16;
    int br = tid >> 2, bc = (tid & 3) * 8;

    v8f acc[2][2] = {};

#define STAGE(buf, k0)                                                        \
    do {                                                                      \
        *(v8u*)&a_lds[buf][ar][ac] =                                          \
            *(const v8u*)&A[(size_t)(m0 + ar) * CC + (k0) + ac];              \
        *(v8u*)&a_lds[buf][ar][ac + 8] =                                      \
            *(const v8u*)&A[(size_t)(m0 + ar) * CC + (k0) + ac + 8];          \
        *(v8u*)&b_lds[buf][br][bc] =                                          \
            *(const v8u*)&Bt[(size_t)(n0 + br) * CC + (k0) + bc];             \
    } while (0)

    STAGE(0, 0);
    __syncthreads();
#pragma unroll
    for (int kb = 0; kb < CC / 32; ++kb) {
        int cur = kb & 1;
        if (kb < CC / 32 - 1) STAGE(cur ^ 1, (kb + 1) * 32);
        v16bf a0 = load_a_frag(&a_lds[cur][mt2 * 32][0], 40);
        v16bf a1 = load_a_frag(&a_lds[cur][mt2 * 32 + 16][0], 40);
        v16bf b0 = load_b_frag(&b_lds[cur][nb * 32][0], 40);
        v16bf b1 = load_b_frag(&b_lds[cur][nb * 32 + 16][0], 40);
        acc[0][0] = wmma_bf16(a0, b0, acc[0][0]);
        acc[0][1] = wmma_bf16(a0, b1, acc[0][1]);
        acc[1][0] = wmma_bf16(a1, b0, acc[1][0]);
        acc[1][1] = wmma_bf16(a1, b1, acc[1][1]);
        __syncthreads();
    }
#undef STAGE

#pragma unroll
    for (int im = 0; im < 2; ++im) {
        int rbase = m0 + mt2 * 32 + im * 16 + ((lane >> 4) << 3);
        int bidx = rbase >> 12;
        int ntok = rbase & (NTOK - 1);
#pragma unroll
        for (int in = 0; in < 2; ++in) {
            int c = n0 + nb * 32 + in * 16 + (lane & 15);
            float bz = bias[c];
            if (MODE == 0) {
#pragma unroll
                for (int r = 0; r < 8; ++r)
                    out_b[(size_t)(rbase + r) * CC + c] =
                        f32_to_bf16((acc[im][in][r] + bz) * alpha);
            } else if (MODE == 1) {
                v8u pk;
#pragma unroll
                for (int r = 0; r < 8; ++r)
                    pk[r] = f32_to_bf16(acc[im][in][r] + bz);
                *(v8u*)(out_b + ((size_t)(bidx * CC + c)) * NTOK + ntok) = pk;
            } else {
                size_t idx = ((size_t)(bidx * CC + c)) * NTOK + ntok;
                v8f rv = *(const v8f*)(resid + idx);
                v8f ov;
#pragma unroll
                for (int r = 0; r < 8; ++r) ov[r] = acc[im][in][r] + bz + rv[r];
                *(v8f*)(out_f + idx) = ov;
            }
        }
    }
}

// ---------------------------------------------------------------------------
// Kernel 4: flash attention. Block = (batch b, 32 query rows i0..i0+31).
// q_t/k_t: [B][N][C] bf16 (q pre-scaled by C^-1/2). v_ct: [B][C][N] bf16.
// q tile staged once into LDS (reused across all 64 key tiles); k/v fragments
// fetched straight from the 192 MB L2 (whole q/k/v set = 48 MB is resident).
// V fragments for a K-step are prefetched before the 8-WMMA chain.
// Wave w owns output channels [w*64, +64).
// ---------------------------------------------------------------------------
__global__ __launch_bounds__(256) void attn_kernel(const u16* __restrict__ q_t,
                                                   const u16* __restrict__ k_t,
                                                   const u16* __restrict__ v_ct,
                                                   u16* __restrict__ ho_t) {
    __shared__ u16 q_lds[32][520];
    __shared__ float s_lds[32][65];
    __shared__ u16 p_lds[32][80];
    __shared__ float red_mx[32][8];
    __shared__ float red_sm[32][8];
    __shared__ float m_run[32], l_run[32], s_alpha[32];

    int b = blockIdx.y;
    int i0 = blockIdx.x * 32;
    int tid = threadIdx.x;
    int w = tid >> 5, lane = tid & 31;
    const u16* qb = q_t + (size_t)b * NTOK * CC;
    const u16* kb = k_t + (size_t)b * NTOK * CC;
    const u16* vb = v_ct + (size_t)b * CC * NTOK;

    // stage q tile [32 x 512] into LDS (8 x 16B per thread, coalesced)
    {
        int qr = tid >> 3, qs = (tid & 7) * 8;
        const u16* src = qb + (size_t)(i0 + qr) * CC;
#pragma unroll
        for (int it = 0; it < 8; ++it)
            *(v8u*)&q_lds[qr][it * 64 + qs] = *(const v8u*)(src + it * 64 + qs);
    }
    if (tid < 32) {
        m_run[tid] = -3.0e38f;
        l_run[tid] = 0.f;
    }
    v8f O[2][4] = {};

    int mt = w >> 2, nt = w & 3;  // this wave's S tile (rows mt*16, cols nt*16)
    int srow = mt * 16 + ((lane >> 4) << 3);
    int scol = nt * 16 + (lane & 15);
    int row = tid >> 3, seg = tid & 7;  // softmax: 8 threads per S row
    __syncthreads();

    for (int j0 = 0; j0 < NTOK; j0 += 64) {
        // ---- S = q . k^T for [32 x 64], K = 512 (16 WMMAs per wave) ----
        v8f s = {};
        const u16* krow = kb + (size_t)(j0 + nt * 16) * CC;
#pragma unroll
        for (int kc = 0; kc < CC; kc += 32) {
            v16bf af = load_a_frag(&q_lds[mt * 16][kc], 520);
            v16bf bfr = load_b_frag(krow + kc, CC);
            s = wmma_bf16(af, bfr, s);
        }
#pragma unroll
        for (int r = 0; r < 8; ++r) s_lds[srow + r][scol] = s[r];
        __syncthreads();

        // ---- online softmax: tile max ----
        float mx = -3.0e38f;
#pragma unroll
        for (int t = 0; t < 8; ++t) mx = fmaxf(mx, s_lds[row][seg * 8 + t]);
        red_mx[row][seg] = mx;
        __syncthreads();
        if (tid < 32) {
            float tm = red_mx[tid][0];
#pragma unroll
            for (int t = 1; t < 8; ++t) tm = fmaxf(tm, red_mx[tid][t]);
            float mnew = fmaxf(m_run[tid], tm);
            s_alpha[tid] = __expf(m_run[tid] - mnew);
            m_run[tid] = mnew;
        }
        __syncthreads();

        // ---- P = exp(S - m), partial row sums, pack bf16 ----
        float mnew = m_run[row];
        float ssum = 0.f;
#pragma unroll
        for (int t = 0; t < 8; ++t) {
            float p = __expf(s_lds[row][seg * 8 + t] - mnew);
            ssum += p;
            p_lds[row][seg * 8 + t] = f32_to_bf16(p);
        }
        red_sm[row][seg] = ssum;
        __syncthreads();
        if (tid < 32) {
            float acc = 0.f;
#pragma unroll
            for (int t = 0; t < 8; ++t) acc += red_sm[tid][t];
            l_run[tid] = l_run[tid] * s_alpha[tid] + acc;
        }

        // ---- rescale O by alpha ----
#pragma unroll
        for (int ih = 0; ih < 2; ++ih) {
            int rb = ih * 16 + ((lane >> 4) << 3);
            float av[8];
#pragma unroll
            for (int r = 0; r < 8; ++r) av[r] = s_alpha[rb + r];
#pragma unroll
            for (int ct = 0; ct < 4; ++ct)
#pragma unroll
                for (int r = 0; r < 8; ++r) O[ih][ct][r] *= av[r];
        }
        // ---- O += P . V : prefetch all 4 V frags per K-step, then 8 WMMAs --
#pragma unroll
        for (int ks = 0; ks < 2; ++ks) {
            v16bf pa0 = load_a_frag(&p_lds[0][ks * 32], 80);
            v16bf pa1 = load_a_frag(&p_lds[16][ks * 32], 80);
            v16bf vf[4];
#pragma unroll
            for (int ct = 0; ct < 4; ++ct)
                vf[ct] = load_b_frag(
                    vb + (size_t)(w * 64 + ct * 16) * NTOK + j0 + ks * 32, NTOK);
#pragma unroll
            for (int ct = 0; ct < 4; ++ct) {
                O[0][ct] = wmma_bf16(pa0, vf[ct], O[0][ct]);
                O[1][ct] = wmma_bf16(pa1, vf[ct], O[1][ct]);
            }
        }
        __syncthreads();
    }

    // ---- finalize: O /= l, store ho_t token-major [B][N][C] bf16 ----
#pragma unroll
    for (int ih = 0; ih < 2; ++ih) {
        int rb = ih * 16 + ((lane >> 4) << 3);
        float li[8];
#pragma unroll
        for (int r = 0; r < 8; ++r) li[r] = 1.f / l_run[rb + r];
#pragma unroll
        for (int ct = 0; ct < 4; ++ct) {
            int c = w * 64 + ct * 16 + (lane & 15);
#pragma unroll
            for (int r = 0; r < 8; ++r) {
                ho_t[(size_t)(b * NTOK + i0 + rb + r) * CC + c] =
                    f32_to_bf16(O[ih][ct][r] * li[r]);
            }
        }
    }
}

// ---------------------------------------------------------------------------
// Host-side launch
// ---------------------------------------------------------------------------
extern "C" void kernel_launch(void* const* d_in, const int* in_sizes, int n_in,
                              void* d_out, int out_size, void* d_ws,
                              size_t ws_size, hipStream_t stream) {
    const float* x = (const float*)d_in[0];
    const float* gn_scale = (const float*)d_in[1];
    const float* gn_bias = (const float*)d_in[2];
    const float* wq = (const float*)d_in[3];
    const float* bq = (const float*)d_in[4];
    const float* wk = (const float*)d_in[5];
    const float* bk = (const float*)d_in[6];
    const float* wv = (const float*)d_in[7];
    const float* bv = (const float*)d_in[8];
    const float* wo = (const float*)d_in[9];
    const float* bo = (const float*)d_in[10];
    float* out = (float*)d_out;

    const size_t ACT = (size_t)MTOT * CC;  // 8M bf16 elems = 16 MB
    const size_t WSZ = (size_t)CC * CC;    // 256K bf16 elems
    u16* h_t = (u16*)d_ws;
    u16* q_t = h_t + ACT;
    u16* k_t = q_t + ACT;
    u16* v_ct = k_t + ACT;
    u16* ho_t = v_ct + ACT;
    u16* wq_b = ho_t + ACT;
    u16* wk_b = wq_b + WSZ;
    u16* wv_b = wk_b + WSZ;
    u16* wo_b = wv_b + WSZ;

    // 1) bf16 weights
    cvt_kernel<<<(int)(WSZ / 256), 256, 0, stream>>>(wq, wq_b, (int)WSZ);
    cvt_kernel<<<(int)(WSZ / 256), 256, 0, stream>>>(wk, wk_b, (int)WSZ);
    cvt_kernel<<<(int)(WSZ / 256), 256, 0, stream>>>(wv, wv_b, (int)WSZ);
    cvt_kernel<<<(int)(WSZ / 256), 256, 0, stream>>>(wo, wo_b, (int)WSZ);

    // 2) GroupNorm -> h_t (token-major bf16)
    gn_kernel<<<BATCH * 32, 256, 0, stream>>>(x, gn_scale, gn_bias, h_t);

    // 3) QKV projections (q pre-scaled by C^-1/2; v stored transposed [C][N])
    dim3 ggrid(MTOT / 128, CC / 64);
    const float qscale = 0.04419417382415922f;  // 512^-0.5
    gemm_tn_kernel<0><<<ggrid, 256, 0, stream>>>(h_t, wq_b, bq, nullptr, qscale,
                                                 q_t, nullptr);
    gemm_tn_kernel<0><<<ggrid, 256, 0, stream>>>(h_t, wk_b, bk, nullptr, 1.f,
                                                 k_t, nullptr);
    gemm_tn_kernel<1><<<ggrid, 256, 0, stream>>>(h_t, wv_b, bv, nullptr, 1.f,
                                                 v_ct, nullptr);

    // 4) flash attention -> ho_t (token-major bf16)
    attn_kernel<<<dim3(NTOK / 32, BATCH), 256, 0, stream>>>(q_t, k_t, v_ct,
                                                            ho_t);

    // 5) output projection + bias + residual -> d_out (f32, channel-major)
    gemm_tn_kernel<2><<<ggrid, 256, 0, stream>>>(ho_t, wo_b, bo, x, 1.f,
                                                 nullptr, out);
}
